// MotionDecoder_50886772523417
// MI455X (gfx1250) — matrix operand order repaired and checked by reading, not compiled, not run on hardware
//
// MotionDecoder (FC + 50-step LSTM decoder + output projection) for MI455X / gfx1250.
// bf16 WMMA (v_wmma_f32_16x16x32_bf16) everywhere, fused w_ih+w_hh, fused
// gate-GEMM + LSTM cell, per-step output projection (no 200MB hs buffer).
// v5: no-wrap software pipeline over padded buffers -- the final prefetch reads
//     one K-step past the logical end (dead registers, in-bounds via +4KB pads),
//     which restores affine addressing so LLVM strength-reduces the loop to
//     immediate-offset load clauses + WMMAs with minimal per-iteration VALU.

#include <hip/hip_runtime.h>
#include <cstdint>

#define BATCH   4096
#define LATENT  64
#define HID     512
#define OUTD    156
#define SEQL    50
#define GATES   (4 * HID)        // 2048
#define KT      (HID / 32)       // 16 K-tiles (K=32 per WMMA)
#define NT_WF   (GATES / 16)     // 128 N-tiles in fused gate weight
#define NT_WO   10               // ceil(156/16) N-tiles in w_out
#define FRAG_E  512              // u16 elements per packed fragment (32 lanes x 16)
#define PAD_B   4096             // overshoot pad (bytes) for pipelined prefetch

typedef __attribute__((ext_vector_type(16))) __bf16 v16bf;
typedef __attribute__((ext_vector_type(8)))  float  v8f;

struct F32B { uint4 lo, hi; };   // 32 bytes == v16bf
static_assert(sizeof(F32B) == sizeof(v16bf), "frag size mismatch");

__device__ __forceinline__ unsigned short f2bf(float f) {
    union { float f; unsigned int u; } x; x.f = f;
    unsigned int r = x.u + 0x7fffu + ((x.u >> 16) & 1u);   // round-to-nearest-even
    return (unsigned short)(r >> 16);
}

// Branch-free activations: v_exp_f32 + v_rcp_f32 only (no libm slow paths).
__device__ __forceinline__ float fast_sigmoid(float x) {
    return __builtin_amdgcn_rcpf(1.0f + __expf(-x));
}
__device__ __forceinline__ float fast_tanh(float x) {
    // tanh(x) = 2*sigmoid(2x) - 1; saturates correctly at both infinities.
    return fmaf(2.0f, __builtin_amdgcn_rcpf(1.0f + __expf(-2.0f * x)), -1.0f);
}

// Packed B-fragment load: 32 contiguous bytes per lane.
__device__ __forceinline__ v16bf load_bfrag(const unsigned short* __restrict__ p) {
    F32B f;
    f.lo = *(const uint4*)(p);
    f.hi = *(const uint4*)(p + 8);
    return __builtin_bit_cast(v16bf, f);
}

// A-fragment chunk pair from a per-lane row pointer: 16B at p, 16B at p+32B.
// (Row-major A, ISA layout: lane<16 holds K = k0+{0..7, 16..23},
//  lane>=16 holds K = k0+{8..15, 24..31}; the lane-dependent +8 is baked into p.)
__device__ __forceinline__ v16bf load_apair(const unsigned short* __restrict__ p) {
    F32B f;
    f.lo = *(const uint4*)(p);
    f.hi = *(const uint4*)(p + 16);
    return __builtin_bit_cast(v16bf, f);
}

// ---- Prep: pack (w_ih + w_hh) -> bf16 B-fragments. One thread = one lane of one frag.
__global__ __launch_bounds__(256) void prep_wf_kernel(
    const float* __restrict__ w_ih, const float* __restrict__ w_hh,
    unsigned short* __restrict__ wf) {
    int idx  = blockIdx.x * 256 + threadIdx.x;      // 128*16*32 total
    int lane = idx & 31;
    int frag = idx >> 5;                            // ntile*KT + ktile
    int ktile = frag % KT, ntile = frag / KT;
    if (ntile >= NT_WF) return;
    int n  = ntile * 16 + (lane & 15);
    int kb = ktile * 32 + ((lane < 16) ? 0 : 8);
    unsigned short* dst = wf + (size_t)frag * FRAG_E + lane * 16;
    const float* ri = w_ih + (size_t)n * HID;
    const float* rh = w_hh + (size_t)n * HID;
#pragma unroll
    for (int j = 0; j < 16; ++j)
        dst[j] = f2bf(ri[kb + (j < 8 ? j : j + 8)] + rh[kb + (j < 8 ? j : j + 8)]);
}

// ---- Prep: pack w_out [156 x 512] -> bf16 B-fragments, zero-padded to N=160.
__global__ __launch_bounds__(256) void prep_wo_kernel(
    const float* __restrict__ w_out, unsigned short* __restrict__ wo) {
    int idx  = blockIdx.x * 256 + threadIdx.x;      // 10*16*32 total
    int lane = idx & 31;
    int frag = idx >> 5;
    int ktile = frag % KT, ntile = frag / KT;
    if (ntile >= NT_WO) return;
    int n  = ntile * 16 + (lane & 15);
    int kb = ktile * 32 + ((lane < 16) ? 0 : 8);
    unsigned short* dst = wo + (size_t)frag * FRAG_E + lane * 16;
#pragma unroll
    for (int j = 0; j < 16; ++j)
        dst[j] = (n < OUTD) ? f2bf(w_out[(size_t)n * HID + kb + (j < 8 ? j : j + 8)]) : (unsigned short)0;
}

// ---- Prep: fused gate bias (f32).
__global__ __launch_bounds__(256) void prep_bias_kernel(
    const float* __restrict__ b_ih, const float* __restrict__ b_hh,
    float* __restrict__ bias) {
    int n = blockIdx.x * 256 + threadIdx.x;
    if (n < GATES) bias[n] = b_ih[n] + b_hh[n];
}

// ---- h0 = latent @ fc_w^T + fc_b  (tiny; plain VALU), stored row-major bf16.
__global__ __launch_bounds__(256) void h0_kernel(
    const float* __restrict__ latent, const float* __restrict__ fc_w,
    const float* __restrict__ fc_b, unsigned short* __restrict__ h0) {
    __shared__ float lat[LATENT];
    int b = blockIdx.x, tid = threadIdx.x;
    if (tid < LATENT) lat[tid] = latent[(size_t)b * LATENT + tid];
    __syncthreads();
    for (int c = tid; c < HID; c += 256) {
        const float* w = fc_w + (size_t)c * LATENT;
        float s = fc_b[c];
#pragma unroll 8
        for (int k = 0; k < LATENT; ++k) s += lat[k] * w[k];
        h0[(size_t)b * HID + c] = f2bf(s);
    }
}

// ---- Fused gate-GEMM + LSTM cell.
// Block = 256 threads = 8 waves; wave tile = 32(M batch) x 16(N hidden) x 4 gates.
// grid = (BATCH/128, HID/32). One set of 4 B-fragments feeds 8 WMMAs per K-step.
// Ping-pong pipeline, no wrap: final prefetch overshoots into buffer padding.
__global__ __launch_bounds__(256) void lstm_step_kernel(
    const unsigned short* __restrict__ hprev, unsigned short* __restrict__ hnext,
    float* __restrict__ c, const unsigned short* __restrict__ wf,
    const float* __restrict__ bias, int t) {
    int lane = threadIdx.x & 31;
    int wid  = threadIdx.x >> 5;
    int m0 = blockIdx.x * 128 + (wid & 3) * 32;   // two 16-row M tiles
    int n0 = blockIdx.y * 32 + (wid >> 2) * 16;
    int ntile = n0 >> 4;

    // Per-lane base pointers, hoisted out of the loop (affine in kk thereafter).
    const unsigned short* pa0 =
        hprev + (size_t)(m0 + (lane & 15)) * HID + ((lane >> 4) << 3);
    const unsigned short* pa1 = pa0 + 16 * HID;
    const unsigned short* wb[4];
#pragma unroll
    for (int g = 0; g < 4; ++g)
        wb[g] = wf + (size_t)((g * 32 + ntile) * KT) * FRAG_E + (size_t)lane * 16;

    v8f acc[2][4];
#pragma unroll
    for (int mi = 0; mi < 2; ++mi)
#pragma unroll
        for (int g = 0; g < 4; ++g)
            acc[mi][g] = (v8f){0.f,0.f,0.f,0.f,0.f,0.f,0.f,0.f};

    v16bf A0[2], A1[2], Bf[4][2];

#define LSTM_LOAD(s, k)                                                        \
    do {                                                                       \
        A0[s]    = load_apair(pa0   + (size_t)(k) * 32);                       \
        A1[s]    = load_apair(pa1   + (size_t)(k) * 32);                       \
        Bf[0][s] = load_bfrag(wb[0] + (size_t)(k) * FRAG_E);                   \
        Bf[1][s] = load_bfrag(wb[1] + (size_t)(k) * FRAG_E);                   \
        Bf[2][s] = load_bfrag(wb[2] + (size_t)(k) * FRAG_E);                   \
        Bf[3][s] = load_bfrag(wb[3] + (size_t)(k) * FRAG_E);                   \
    } while (0)

#define LSTM_WMMA(s)                                                                                              \
    do {                                                                                                          \
        acc[0][0] = __builtin_amdgcn_wmma_f32_16x16x32_bf16(false, A0[s], false, Bf[0][s], (short)0, acc[0][0], false, false); \
        acc[1][0] = __builtin_amdgcn_wmma_f32_16x16x32_bf16(false, A1[s], false, Bf[0][s], (short)0, acc[1][0], false, false); \
        acc[0][1] = __builtin_amdgcn_wmma_f32_16x16x32_bf16(false, A0[s], false, Bf[1][s], (short)0, acc[0][1], false, false); \
        acc[1][1] = __builtin_amdgcn_wmma_f32_16x16x32_bf16(false, A1[s], false, Bf[1][s], (short)0, acc[1][1], false, false); \
        acc[0][2] = __builtin_amdgcn_wmma_f32_16x16x32_bf16(false, A0[s], false, Bf[2][s], (short)0, acc[0][2], false, false); \
        acc[1][2] = __builtin_amdgcn_wmma_f32_16x16x32_bf16(false, A1[s], false, Bf[2][s], (short)0, acc[1][2], false, false); \
        acc[0][3] = __builtin_amdgcn_wmma_f32_16x16x32_bf16(false, A0[s], false, Bf[3][s], (short)0, acc[0][3], false, false); \
        acc[1][3] = __builtin_amdgcn_wmma_f32_16x16x32_bf16(false, A1[s], false, Bf[3][s], (short)0, acc[1][3], false, false); \
    } while (0)

    LSTM_LOAD(0, 0);                       // prologue
#pragma unroll 1
    for (int kk = 0; kk < KT; kk += 2) {
        LSTM_LOAD(1, kk + 1);              // prefetch odd K-step
        LSTM_WMMA(0);                      // compute even K-step
        LSTM_LOAD(0, kk + 2);              // prefetch next even (kk+2==KT overshoots
        LSTM_WMMA(1);                      //   into padding; result is never consumed)
    }
#undef LSTM_LOAD
#undef LSTM_WMMA

    // C/D layout: lane's 8 values share N = n0 + (lane&15); M = (v + 8*(lane>=16)) + mtile.
    int n = n0 + (lane & 15);
    float bi = bias[0 * HID + n], bf = bias[1 * HID + n];
    float bg = bias[2 * HID + n], bo = bias[3 * HID + n];
#pragma unroll
    for (int mi = 0; mi < 2; ++mi) {
        int mbase = m0 + mi * 16 + ((lane >> 4) << 3);
#pragma unroll
        for (int v = 0; v < 8; ++v) {
            int m = mbase + v;
            float iv = fast_sigmoid(acc[mi][0][v] + bi);
            float fv = fast_sigmoid(acc[mi][1][v] + bf);
            float gv = fast_tanh   (acc[mi][2][v] + bg);
            float ov = fast_sigmoid(acc[mi][3][v] + bo);
            size_t ci = (size_t)m * HID + n;
            float cold = t ? c[ci] : 0.0f;       // c0 == 0: skip load at t=0
            float cn = fv * cold + iv * gv;
            c[ci] = cn;
            hnext[ci] = f2bf(ov * fast_tanh(cn));
        }
    }
}

// ---- Per-step output projection: out[b][t][:] = h[b] @ w_out^T + b_out.
// grid = (BATCH/512, NT_WO); wave tile = 64(M) x 16(N): one B-frag feeds 4 WMMAs.
// Same no-wrap ping-pong pipeline.
__global__ __launch_bounds__(256) void out_proj_kernel(
    const unsigned short* __restrict__ h, const unsigned short* __restrict__ wo,
    const float* __restrict__ b_out, float* __restrict__ out, int t) {
    int lane = threadIdx.x & 31;
    int wid  = threadIdx.x >> 5;
    int m0 = blockIdx.x * 512 + wid * 64;
    int ntile = blockIdx.y;
    int n0 = ntile * 16;

    const unsigned short* wl = wo + (size_t)(ntile * KT) * FRAG_E + (size_t)lane * 16;
    const unsigned short* pa[4];
#pragma unroll
    for (int mi = 0; mi < 4; ++mi)
        pa[mi] = h + (size_t)(m0 + mi * 16 + (lane & 15)) * HID + ((lane >> 4) << 3);

    v8f acc[4];
#pragma unroll
    for (int mi = 0; mi < 4; ++mi)
        acc[mi] = (v8f){0.f,0.f,0.f,0.f,0.f,0.f,0.f,0.f};

    v16bf A[4][2], Bv[2];

#define PROJ_LOAD(s, k)                                                        \
    do {                                                                       \
        Bv[s]   = load_bfrag(wl    + (size_t)(k) * FRAG_E);                    \
        A[0][s] = load_apair(pa[0] + (size_t)(k) * 32);                        \
        A[1][s] = load_apair(pa[1] + (size_t)(k) * 32);                        \
        A[2][s] = load_apair(pa[2] + (size_t)(k) * 32);                        \
        A[3][s] = load_apair(pa[3] + (size_t)(k) * 32);                        \
    } while (0)

#define PROJ_WMMA(s)                                                                                          \
    do {                                                                                                      \
        acc[0] = __builtin_amdgcn_wmma_f32_16x16x32_bf16(false, A[0][s], false, Bv[s], (short)0, acc[0], false, false); \
        acc[1] = __builtin_amdgcn_wmma_f32_16x16x32_bf16(false, A[1][s], false, Bv[s], (short)0, acc[1], false, false); \
        acc[2] = __builtin_amdgcn_wmma_f32_16x16x32_bf16(false, A[2][s], false, Bv[s], (short)0, acc[2], false, false); \
        acc[3] = __builtin_amdgcn_wmma_f32_16x16x32_bf16(false, A[3][s], false, Bv[s], (short)0, acc[3], false, false); \
    } while (0)

    PROJ_LOAD(0, 0);
#pragma unroll 1
    for (int kk = 0; kk < KT; kk += 2) {
        PROJ_LOAD(1, kk + 1);
        PROJ_WMMA(0);
        PROJ_LOAD(0, kk + 2);   // final prefetch overshoots into padding (dead)
        PROJ_WMMA(1);
    }
#undef PROJ_LOAD
#undef PROJ_WMMA

    int n = n0 + (lane & 15);
    float bo = (n < OUTD) ? b_out[n] : 0.0f;
#pragma unroll
    for (int mi = 0; mi < 4; ++mi) {
        int mbase = m0 + mi * 16 + ((lane >> 4) << 3);
#pragma unroll
        for (int v = 0; v < 8; ++v) {
            int m = mbase + v;                               // batch index
            if (n < OUTD)
                out[((size_t)m * SEQL + t) * OUTD + n] = acc[mi][v] + bo;
        }
    }
}

extern "C" void kernel_launch(void* const* d_in, const int* in_sizes, int n_in,
                              void* d_out, int out_size, void* d_ws, size_t ws_size,
                              hipStream_t stream) {
    const float* latent = (const float*)d_in[0];
    const float* fc_w   = (const float*)d_in[1];
    const float* fc_b   = (const float*)d_in[2];
    const float* w_ih   = (const float*)d_in[3];
    const float* b_ih   = (const float*)d_in[4];
    const float* w_hh   = (const float*)d_in[5];
    const float* b_hh   = (const float*)d_in[6];
    const float* w_out  = (const float*)d_in[7];
    const float* b_out  = (const float*)d_in[8];
    float* out = (float*)d_out;

    // Workspace carve-up (256B aligned, each buffer padded for prefetch
    // overshoot): ~18.2 MB total.
    char* ws = (char*)d_ws;
    size_t off = 0;
    auto carve = [&](size_t bytes) {
        size_t p = off;
        off += (bytes + PAD_B + 255) & ~(size_t)255;
        return (void*)(ws + p);
    };
    unsigned short* wf   = (unsigned short*)carve((size_t)NT_WF * KT * FRAG_E * 2); // 2 MB
    unsigned short* wo   = (unsigned short*)carve((size_t)NT_WO * KT * FRAG_E * 2); // 160 KB
    float*          bias = (float*)         carve((size_t)GATES * 4);              // 8 KB
    unsigned short* hb0  = (unsigned short*)carve((size_t)BATCH * HID * 2);        // 4 MB
    unsigned short* hb1  = (unsigned short*)carve((size_t)BATCH * HID * 2);        // 4 MB
    float*          cst  = (float*)         carve((size_t)BATCH * HID * 4);        // 8 MB
    (void)ws_size; (void)in_sizes; (void)n_in; (void)out_size;

    // Weight/bias packing (deterministic; rewritten every call).
    prep_wf_kernel  <<<(NT_WF * KT * 32) / 256, 256, 0, stream>>>(w_ih, w_hh, wf);
    prep_wo_kernel  <<<(NT_WO * KT * 32) / 256, 256, 0, stream>>>(w_out, wo);
    prep_bias_kernel<<<GATES / 256,            256, 0, stream>>>(b_ih, b_hh, bias);

    // h0 = latent @ fc_w^T + fc_b  -> hb0 (bf16 row-major).
    h0_kernel<<<BATCH, 256, 0, stream>>>(latent, fc_w, fc_b, hb0);

    unsigned short* hbuf[2] = {hb0, hb1};
    dim3 gStep(BATCH / 128, HID / 32);  // 32 x 16 blocks
    dim3 gProj(BATCH / 512, NT_WO);     // 8 x 10 blocks
    for (int t = 0; t < SEQL; ++t) {
        unsigned short* hp = hbuf[t & 1];
        unsigned short* hn = hbuf[(t + 1) & 1];
        lstm_step_kernel<<<gStep, 256, 0, stream>>>(hp, hn, cst, wf, bias, t);
        out_proj_kernel <<<gProj, 256, 0, stream>>>(hn, wo, b_out, out, t);
    }
}
